// ResidualGNNBlock_37168646979916
// MI455X (gfx1250) — matrix operand order repaired and checked by reading, not compiled, not run on hardware
//
#include <hip/hip_runtime.h>

#define D 128
#define ED 7
#define K1 288          // 2D+ED = 263 padded to 9*32
#define MS_LD 304       // LDS stride for Ms (ushorts; 608B = 16B-aligned rows)
#define HS_LD 144       // LDS stride for Hs (288B rows, 16B aligned)
#define OC 1024         // fused node-side output columns
#define EPS 1e-5f
#define FRAG 512        // ushorts per packed 32x16 B fragment (32 lanes x 16)

typedef __bf16          v16bf __attribute__((ext_vector_type(16)));
typedef unsigned short  v16us __attribute__((ext_vector_type(16)));
typedef float           v8f   __attribute__((ext_vector_type(8)));

static __device__ __forceinline__ unsigned short f2bf(float f) {
    unsigned int u = __float_as_uint(f);
    u += 0x7FFFu + ((u >> 16) & 1u);      // round-to-nearest-even
    return (unsigned short)(u >> 16);
}

static __device__ __forceinline__ float sigmoidf(float v) {
    return 1.0f / (1.0f + __expf(-v));
}

// A fragment: 16x32 bf16 (MxK), ISA 7.12.2 layout. Per lane the elements are two
// contiguous 8-ushort runs -> compiler merges to 2x ds_load_b128.
static __device__ __forceinline__ v16bf load_a_frag(const unsigned short* base, int ldm, int lane) {
    int g = (lane >> 4) & 1;
    int m = lane & 15;
    const unsigned short* row = base + m * ldm;
    v16us t;
#pragma unroll
    for (int v = 0; v < 8; ++v) {
        int kb = (v < 4) ? (g * 8 + 2 * v) : (16 + g * 8 + 2 * (v - 4));
        t[2 * v]     = row[kb];
        t[2 * v + 1] = row[kb + 1];
    }
    return __builtin_bit_cast(v16bf, t);
}

// B fragment from pre-packed fragment-major table: lane's 16 bf16 are contiguous
// (32B) -> 2x global_load_b128, no repacking.
static __device__ __forceinline__ v16bf load_b_packed(const unsigned short* frag, int lane) {
    v16us t = *(const v16us*)(frag + lane * 16);
    return __builtin_bit_cast(v16bf, t);
}

static __device__ __forceinline__ v8f wmma_bf16(v16bf a, v16bf b, v8f c) {
    return __builtin_amdgcn_wmma_f32_16x16x32_bf16(false, a, false, b, (short)0, c, false, false);
}

// ---------------------------------------------------------------- prep kernels
__global__ void prep_x_kernel(const float* __restrict__ x, unsigned short* __restrict__ xh, int n) {
    int i = blockIdx.x * blockDim.x + threadIdx.x;
    if (i < n) xh[i] = f2bf(x[i]);
}

// Decompose packed-fragment flat index -> (k, o) for a matrix with NT col-tiles.
// content[frag*512 + lane*16 + j] = W[kt*32 + 8*g + j + (j >= 8 ? 8 : 0)][nt*16 + n]
static __device__ __forceinline__ void frag_decode(int i, int NT, int& k, int& o) {
    int frag = i >> 9;
    int rr   = i & 511;
    int lane = rr >> 4, j = rr & 15;
    int g = lane >> 4, n = lane & 15;
    int kt = frag / NT, nt = frag % NT;
    k = kt * 32 + 8 * g + j + (j >= 8 ? 8 : 0);
    o = nt * 16 + n;
}

__global__ void prep_weights_kernel(const float* __restrict__ W1, const float* __restrict__ W2,
                                    const float* __restrict__ Wg, const float* __restrict__ Wih,
                                    const float* __restrict__ Whh,
                                    unsigned short* __restrict__ W1f,   // 9x8 frags  (K=288, NO=128)
                                    unsigned short* __restrict__ W2f,   // 4x8 frags  (K=128, NO=128)
                                    unsigned short* __restrict__ Wpf)   // 4x64 frags (K=128, NO=1024)
{
    int i = blockIdx.x * blockDim.x + threadIdx.x;
    if (i < K1 * D) {          // W1f: W1[o][k], zero pad k >= 263
        int k, o; frag_decode(i, D / 16, k, o);
        float v = (k < 2 * D + ED) ? W1[(size_t)o * (2 * D + ED) + k] : 0.0f;
        W1f[i] = f2bf(v);
    }
    if (i < D * D) {           // W2f: W2[o][k]
        int k, o; frag_decode(i, D / 16, k, o);
        W2f[i] = f2bf(W2[(size_t)o * D + k]);
    }
    if (i < D * OC) {          // Wpf: o: 0..383 Wih | 384..767 Whh | 768..895 Wg_x | 896..1023 Wg_a
        int k, o; frag_decode(i, OC / 16, k, o);
        float v;
        if (o < 384)      v = Wih[(size_t)o * D + k];
        else if (o < 768) v = Whh[(size_t)(o - 384) * D + k];
        else if (o < 896) { int oo = o - 768; v = Wg[(size_t)oo * 384 + k] + Wg[(size_t)oo * 384 + 256 + k]; }
        else              { int oo = o - 896; v = Wg[(size_t)oo * 384 + 128 + k]; }
        Wpf[i] = f2bf(v);
    }
}

// ------------------------------------------------------- edge message + scatter
__global__ __launch_bounds__(256) void edge_msg_kernel(
    const unsigned short* __restrict__ xh,       // [N][128] bf16
    const int* __restrict__ edge_index,          // [2][E]
    const float* __restrict__ edge_attr,         // [E][7]
    const unsigned short* __restrict__ W1f,      // packed frags
    const float* __restrict__ b1,
    const unsigned short* __restrict__ W2f,      // packed frags
    const float* __restrict__ b2,
    float* __restrict__ aggr,                    // [N][128]
    int nEdges, int nNodes)
{
    __shared__ __align__(16) unsigned short Ms[16][MS_LD];
    __shared__ __align__(16) unsigned short Hs[16][HS_LD];
    __shared__ int sSrc[16], sDst[16];

    const int tid  = threadIdx.x;
    const int lane = tid & 31;
    const int wave = tid >> 5;                   // 8 waves
    const int e0   = blockIdx.x * 16;

    if (tid < 16) {
        int e = e0 + tid;
        int s = 0, d = -1;
        if (e < nEdges) {
            s = edge_index[e];
            d = edge_index[nEdges + e];
            if (s < 0) s = 0; if (s >= nNodes) s = nNodes - 1;
            if (d < 0) d = 0; if (d >= nNodes) d = nNodes - 1;
        }
        sSrc[tid] = s;
        sDst[tid] = d;
    }
    __syncthreads();

    // vectorized gather: thread (r = tid>>4, seg = tid&15) moves 8 contiguous bf16
    {
        int r = tid >> 4, seg = tid & 15;
        uint4 vi = make_uint4(0, 0, 0, 0), vj = vi;
        if (e0 + r < nEdges) {
            vi = ((const uint4*)(xh + (size_t)sDst[r] * D))[seg];
            vj = ((const uint4*)(xh + (size_t)sSrc[r] * D))[seg];
        }
        *(uint4*)&Ms[r][seg * 8]     = vi;      // x_i -> cols 0..127
        *(uint4*)&Ms[r][D + seg * 8] = vj;      // x_j -> cols 128..255
    }
    // edge_attr -> cols 256..262, zero-pad 263..287
    for (int i = tid; i < 16 * 32; i += 256) {
        int r = i >> 5, c = i & 31;
        float v = 0.0f;
        if (e0 + r < nEdges && c < ED) v = edge_attr[(size_t)(e0 + r) * ED + c];
        Ms[r][256 + c] = f2bf(v);
    }
    __syncthreads();

    const int ob = wave * 16;                    // this wave's 16 output cols

    // layer 1: h = relu(msg_in @ W1^T + b1), K = 288 (9 WMMAs)
    v8f c1 = {};
#pragma unroll
    for (int kt = 0; kt < 9; ++kt) {
        v16bf a = load_a_frag(&Ms[0][kt * 32], MS_LD, lane);
        v16bf b = load_b_packed(W1f + (size_t)(kt * 8 + wave) * FRAG, lane);
        c1 = wmma_bf16(a, b, c1);
    }
    {
        int g = (lane >> 4) & 1, n = lane & 15;
        float bb = b1[ob + n];
#pragma unroll
        for (int v = 0; v < 8; ++v) {
            float h = c1[v] + bb;
            h = h > 0.0f ? h : 0.0f;
            Hs[v + 8 * g][ob + n] = f2bf(h);
        }
    }
    __syncthreads();

    // layer 2: msg = h @ W2^T + b2, K = 128 (4 WMMAs), then scatter-add
    v8f c2 = {};
#pragma unroll
    for (int kt = 0; kt < 4; ++kt) {
        v16bf a = load_a_frag(&Hs[0][kt * 32], HS_LD, lane);
        v16bf b = load_b_packed(W2f + (size_t)(kt * 8 + wave) * FRAG, lane);
        c2 = wmma_bf16(a, b, c2);
    }
    {
        int g = (lane >> 4) & 1, n = lane & 15;
        float bb = b2[ob + n];
#pragma unroll
        for (int v = 0; v < 8; ++v) {
            int m = v + 8 * g;
            int d = sDst[m];
            if (d >= 0) atomicAdd(&aggr[(size_t)d * D + ob + n], c2[v] + bb);
        }
    }
}

// ------------------------------------------------- node update (GEMM + GRU + LN)
__global__ __launch_bounds__(256) void node_update_kernel(
    const float* __restrict__ x,
    const unsigned short* __restrict__ xh,
    const float* __restrict__ aggr,
    const unsigned short* __restrict__ Wpf,      // packed frags [4 ktiles][64 ntiles]
    const float* __restrict__ bih, const float* __restrict__ bhh,
    const float* __restrict__ bg,
    const float* __restrict__ ln1g, const float* __restrict__ ln1b,
    const float* __restrict__ ln2g, const float* __restrict__ ln2b,
    float* __restrict__ out, int nNodes)
{
    __shared__ __align__(16) unsigned short Xs[16][HS_LD];
    __shared__ __align__(16) unsigned short As[16][HS_LD];
    __shared__ float acc[16][OC];                // 64 KB
    __shared__ float red1[16][17];
    __shared__ float red2[16][17];

    const int tid  = threadIdx.x;
    const int lane = tid & 31;
    const int wave = tid >> 5;
    const int n0   = blockIdx.x * 16;

    {   // vectorized tile load: thread (r, seg) moves 8 contiguous elements
        int r = tid >> 4, seg = tid & 15;
        int node = n0 + r;
        uint4 xv = make_uint4(0, 0, 0, 0);
        uint4 av = xv;
        if (node < nNodes) {
            xv = ((const uint4*)(xh + (size_t)node * D))[seg];
            const float4* pa = (const float4*)(aggr + (size_t)node * D) + seg * 2;
            float4 a0 = pa[0], a1 = pa[1];
            av.x = (unsigned)f2bf(a0.x) | ((unsigned)f2bf(a0.y) << 16);
            av.y = (unsigned)f2bf(a0.z) | ((unsigned)f2bf(a0.w) << 16);
            av.z = (unsigned)f2bf(a1.x) | ((unsigned)f2bf(a1.y) << 16);
            av.w = (unsigned)f2bf(a1.z) | ((unsigned)f2bf(a1.w) << 16);
        }
        *(uint4*)&Xs[r][seg * 8] = xv;
        *(uint4*)&As[r][seg * 8] = av;
    }
    __syncthreads();

    // 64 output tiles of 16 cols; A = aggr for gi (t<24) and Wg_a (t>=56), else x
    for (int t = wave; t < 64; t += 8) {
        unsigned short (*Asrc)[HS_LD] = (t < 24 || t >= 56) ? As : Xs;
        v8f c = {};
#pragma unroll
        for (int kt = 0; kt < 4; ++kt) {
            v16bf a = load_a_frag(&Asrc[0][kt * 32], HS_LD, lane);
            v16bf b = load_b_packed(Wpf + (size_t)(kt * 64 + t) * FRAG, lane);
            c = wmma_bf16(a, b, c);
        }
        int g = (lane >> 4) & 1, n = lane & 15;
#pragma unroll
        for (int v = 0; v < 8; ++v) acc[v + 8 * g][t * 16 + n] = c[v];
    }
    __syncthreads();

    // epilogue: thread (r = tid>>4, c0 = tid&15) owns cols c0+16j of row r
    const int r  = tid >> 4;
    const int c0 = tid & 15;
    const int node = n0 + r;

    float o1[8], xr[8];
    float s1 = 0.0f, s2 = 0.0f;
#pragma unroll
    for (int j = 0; j < 8; ++j) {
        int c = c0 + 16 * j;
        float xv = (node < nNodes) ? x[(size_t)node * D + c] : 0.0f;
        xr[j] = xv;
        float ir  = acc[r][c]       + bih[c];
        float iz  = acc[r][128 + c] + bih[128 + c];
        float in_ = acc[r][256 + c] + bih[256 + c];
        float hr  = acc[r][384 + c] + bhh[c];
        float hz  = acc[r][512 + c] + bhh[128 + c];
        float hn  = acc[r][640 + c] + bhh[256 + c];
        float gate = sigmoidf(acc[r][768 + c] + acc[r][896 + c] + bg[c]);
        float rg = sigmoidf(ir + hr);
        float z  = sigmoidf(iz + hz);
        float nn = tanhf(in_ + rg * hn);
        float upd = (1.0f - z) * nn + z * xv;
        float oo = gate * upd + (1.0f - gate) * xv;
        o1[j] = oo;
        s1 += oo; s2 += oo * oo;
    }
    red1[r][c0] = s1; red2[r][c0] = s2;
    __syncthreads();
    float mu = 0.0f, m2 = 0.0f;
#pragma unroll
    for (int i = 0; i < 16; ++i) { mu += red1[r][i]; m2 += red2[r][i]; }
    mu *= (1.0f / 128.0f);
    float var = m2 * (1.0f / 128.0f) - mu * mu;
    float inv1 = rsqrtf(var + EPS);

    float o2[8];
    float t1 = 0.0f, t2 = 0.0f;
#pragma unroll
    for (int j = 0; j < 8; ++j) {
        int c = c0 + 16 * j;
        float v = (o1[j] - mu) * inv1 * ln1g[c] + ln1b[c] + xr[j];   // LN1 + residual
        o2[j] = v;
        t1 += v; t2 += v * v;
    }
    __syncthreads();
    red1[r][c0] = t1; red2[r][c0] = t2;
    __syncthreads();
    float mu2 = 0.0f, m22 = 0.0f;
#pragma unroll
    for (int i = 0; i < 16; ++i) { mu2 += red1[r][i]; m22 += red2[r][i]; }
    mu2 *= (1.0f / 128.0f);
    float var2 = m22 * (1.0f / 128.0f) - mu2 * mu2;
    float inv2 = rsqrtf(var2 + EPS);

    if (node < nNodes) {
#pragma unroll
        for (int j = 0; j < 8; ++j) {
            int c = c0 + 16 * j;
            out[(size_t)node * D + c] = (o2[j] - mu2) * inv2 * ln2g[c] + ln2b[c];
        }
    }
}

// ------------------------------------------------------------------- launcher
extern "C" void kernel_launch(void* const* d_in, const int* in_sizes, int n_in,
                              void* d_out, int out_size, void* d_ws, size_t ws_size,
                              hipStream_t stream) {
    const float* x          = (const float*)d_in[0];
    const int*   edge_index = (const int*)d_in[1];
    const float* edge_attr  = (const float*)d_in[2];
    const float* W1  = (const float*)d_in[3];
    const float* b1  = (const float*)d_in[4];
    const float* W2  = (const float*)d_in[5];
    const float* b2  = (const float*)d_in[6];
    const float* Wg  = (const float*)d_in[7];
    const float* bg  = (const float*)d_in[8];
    const float* Wih = (const float*)d_in[9];
    const float* Whh = (const float*)d_in[10];
    const float* bih = (const float*)d_in[11];
    const float* bhh = (const float*)d_in[12];
    const float* ln1g = (const float*)d_in[13];
    const float* ln1b = (const float*)d_in[14];
    const float* ln2g = (const float*)d_in[15];
    const float* ln2b = (const float*)d_in[16];

    const int N = in_sizes[0] / D;       // 50000
    const int E = in_sizes[2] / ED;      // 600000

    // workspace carve-out (256B aligned)
    char* ws = (char*)d_ws;
    size_t off = 0;
    auto alloc = [&](size_t bytes) { char* p = ws + off; off = (off + bytes + 255) & ~(size_t)255; return p; };
    float*          aggr = (float*)alloc((size_t)N * D * sizeof(float));
    unsigned short* xh   = (unsigned short*)alloc((size_t)N * D * sizeof(unsigned short));
    unsigned short* W1f  = (unsigned short*)alloc((size_t)K1 * D * sizeof(unsigned short));
    unsigned short* W2f  = (unsigned short*)alloc((size_t)D * D * sizeof(unsigned short));
    unsigned short* Wpf  = (unsigned short*)alloc((size_t)D * OC * sizeof(unsigned short));
    (void)ws_size;

    hipMemsetAsync(aggr, 0, (size_t)N * D * sizeof(float), stream);

    {
        int n = N * D;
        prep_x_kernel<<<(n + 255) / 256, 256, 0, stream>>>(x, xh, n);
    }
    {
        int n = D * OC;  // largest packed table
        prep_weights_kernel<<<(n + 255) / 256, 256, 0, stream>>>(W1, W2, Wg, Wih, Whh, W1f, W2f, Wpf);
    }
    edge_msg_kernel<<<(E + 15) / 16, 256, 0, stream>>>(xh, edge_index, edge_attr,
                                                       W1f, b1, W2f, b2, aggr, E, N);
    node_update_kernel<<<(N + 15) / 16, 256, 0, stream>>>(x, xh, aggr, Wpf,
                                                          bih, bhh, bg,
                                                          ln1g, ln1b, ln2g, ln2b,
                                                          (float*)d_out, N);
}
